// GATLayer_3358664425576
// MI455X (gfx1250) — matrix operand order repaired and checked by reading, not compiled, not run on hardware
//
#include <hip/hip_runtime.h>
#include <hip/hip_bf16.h>

typedef __attribute__((ext_vector_type(16))) _Float16 v16h;
typedef __attribute__((ext_vector_type(2)))  __fp16   v2hf;   // cvt_pkrtz result
typedef __attribute__((ext_vector_type(8)))  float    v8f;

#define N_NODES 6144
#define IN_F    128
#define OUT_F   128
#define H_HEADS 4
#define OPH     32
#define ALPHA   0.2f

// ---- CDNA5 async global->LDS path (guarded; falls back to sync copy) ------
#if defined(__AMDGCN__) && __has_builtin(__builtin_amdgcn_global_load_async_to_lds_b128)
#define USE_ASYNC 1
#else
#define USE_ASYNC 0
#endif

// exact pointee type expected by the builtin: int __vector(4)
typedef int i32x4 __attribute__((vector_size(16)));
typedef __attribute__((address_space(1))) i32x4 gl_i32x4;
typedef __attribute__((address_space(3))) i32x4 ld_i32x4;

__device__ __forceinline__ void wait_async_lds() {
#if USE_ASYNC
#if __has_builtin(__builtin_amdgcn_s_wait_asynccnt)
    __builtin_amdgcn_s_wait_asynccnt(0);
#else
    asm volatile("s_wait_asynccnt 0" ::: "memory");
#endif
#endif
}

// ---------------------------------------------------------------------------
// Phase 1: wh = h_mat @ weight, stored TRANSPOSED as f16 (whT[f][n]) so that
// phase 2 B-fragments are two contiguous 16B runs per lane; plus logits lp[n,h], lc[n,h].
// ---------------------------------------------------------------------------
__global__ __launch_bounds__(IN_F)
void gat_prep(const float* __restrict__ h_mat,
              const float* __restrict__ weight,
              const float* __restrict__ att_weight,
              _Float16*    __restrict__ whT,
              float*       __restrict__ lp,
              float*       __restrict__ lc)
{
    __shared__ float hrow[IN_F];
    const int n = blockIdx.x;
    const int f = threadIdx.x;
    hrow[f] = h_mat[(size_t)n * IN_F + f];
    __syncthreads();

    float acc = 0.f;
    #pragma unroll 8
    for (int k = 0; k < IN_F; ++k)
        acc = fmaf(hrow[k], weight[k * OUT_F + f], acc);

    whT[(size_t)f * N_NODES + n] = (_Float16)acc;   // transposed store

    const int h  = f >> 5;                           // head = wave id (wave32)
    const int fo = f & 31;
    float p = acc * att_weight[h * 2 * OPH + fo];            // a_src
    float c = acc * att_weight[h * 2 * OPH + OPH + fo];      // a_dst
    #pragma unroll
    for (int off = 16; off >= 1; off >>= 1) {
        p += __shfl_down(p, off, 32);
        c += __shfl_down(c, off, 32);
    }
    if (fo == 0) {
        lp[n * H_HEADS + h] = p;
        lc[n * H_HEADS + h] = c;
    }
}

// ---------------------------------------------------------------------------
// Stage one 16x32 adj tile (async if available) + transposed lc chunk.
// 128 threads: one 16B adj vector each, one lc scalar each.
// ---------------------------------------------------------------------------
__device__ __forceinline__ void stage_tile(const float* __restrict__ adj,
                                           const float* __restrict__ lc,
                                           float* s_adj_buf, float* s_lct_buf,
                                           int n0, int j0, int tid)
{
    const int m = tid >> 3, k = (tid & 7) * 4;
    const float* gsrc = adj + (size_t)(n0 + m) * N_NODES + j0 + k;
    float* ldst = s_adj_buf + m * 32 + k;
#if USE_ASYNC
    __builtin_amdgcn_global_load_async_to_lds_b128(
        (gl_i32x4*)(uintptr_t)gsrc,
        (ld_i32x4*)(uint32_t)(uintptr_t)ldst, 0, 0);
#else
    *(float4*)ldst = *(const float4*)gsrc;
#endif
    const int kk = tid >> 2, hh = tid & 3;            // transpose lc -> [h][k]
    s_lct_buf[hh * 32 + kk] = lc[(size_t)(j0 + kk) * H_HEADS + hh];
}

// ---------------------------------------------------------------------------
// Phase 2: fused masked-softmax aggregation, single pass over adj.
// Workgroup = 16 rows x 4 waves (wave h = head h). Per 32-wide j chunk:
// ping-pong-staged adj tile, branchless exp/mask A-build packed with
// v_cvt_pk_rtz_f16_f32, B from transposed whT via 2x global_load_b128/lane,
// 2x v_wmma_f32_16x16x32_f16 into persistent accumulators.
// ---------------------------------------------------------------------------
__global__ __launch_bounds__(128)
void gat_main(const float*    __restrict__ adj,
              const _Float16* __restrict__ whT,
              const float*    __restrict__ lp,
              const float*    __restrict__ lc,
              float*          __restrict__ out)
{
    __shared__ __align__(16) float s_adj[2][16 * 32];     // ping-pong [m][k]
    __shared__ __align__(16) float s_lct[2][H_HEADS * 32];// ping-pong [h][k]
    __shared__               float s_row[H_HEADS][16];    // softmax row sums

    const int tid  = threadIdx.x;
    const int h    = tid >> 5;        // head (wave id)
    const int lane = tid & 31;
    const int M    = lane & 15;       // A row owned by this lane
    const int hi   = lane >> 4;       // lane half -> K subset / C row block
    const int col  = lane & 15;       // B/C column (feature within 16-group)
    const int n0   = blockIdx.x * 16;

    const float lpv = lp[(size_t)(n0 + M) * H_HEADS + h];

    // per-lane B base pointers: whT[(h*32+col(+16)) * N + 8*hi + j0 ...]
    const _Float16* pb0 = whT + (size_t)(h * OPH + col) * N_NODES + 8 * hi;
    const _Float16* pb1 = pb0 + (size_t)16 * N_NODES;

    v8f c0 = {};   // f = 0..15
    v8f c1 = {};   // f = 16..31
    float rowsum = 0.f;

    // prologue: stage tile 0
    stage_tile(adj, lc, s_adj[0], s_lct[0], n0, 0, tid);
    wait_async_lds();
    __syncthreads();

    for (int it = 0; it < N_NODES / 32; ++it) {
        const int j0  = it * 32;
        const int cur = it & 1;

        // software pipeline: kick off next tile before computing this one
        if (j0 + 32 < N_NODES)
            stage_tile(adj, lc, s_adj[cur ^ 1], s_lct[cur ^ 1], n0, j0 + 32, tid);
        if (j0 + 64 < N_NODES) {                    // prefetch tile i+2
            const int m = tid >> 3, k = (tid & 7) * 4;
            __builtin_prefetch(adj + (size_t)(n0 + m) * N_NODES + j0 + 64 + k, 0, 1);
        }

        // ---- A-fragment: lane's 16 k values = two contiguous 8-runs ----
        const float* sa = s_adj[cur] + M * 32;
        const float* sl = s_lct[cur] + h * 32;
        const float4 a0 = *(const float4*)(sa + 8 * hi);
        const float4 a1 = *(const float4*)(sa + 8 * hi + 4);
        const float4 a2 = *(const float4*)(sa + 16 + 8 * hi);
        const float4 a3 = *(const float4*)(sa + 16 + 8 * hi + 4);
        const float4 l0 = *(const float4*)(sl + 8 * hi);
        const float4 l1 = *(const float4*)(sl + 8 * hi + 4);
        const float4 l2 = *(const float4*)(sl + 16 + 8 * hi);
        const float4 l3 = *(const float4*)(sl + 16 + 8 * hi + 4);

        float rs = 0.f;
        auto wfun = [&](float av, float lv) -> float {   // branchless
            float x = lpv + lv;
            x = (x > 0.f) ? x : ALPHA * x;               // leaky_relu
            float e = __expf(x);
            float w = e * ((av > 0.f) ? 1.0f : 0.0f);    // cndmask * mul
            rs += w;
            return w;
        };

        union { v16h v; v2hf p[8]; } A;
        A.p[0] = __builtin_amdgcn_cvt_pkrtz(wfun(a0.x, l0.x), wfun(a0.y, l0.y));
        A.p[1] = __builtin_amdgcn_cvt_pkrtz(wfun(a0.z, l0.z), wfun(a0.w, l0.w));
        A.p[2] = __builtin_amdgcn_cvt_pkrtz(wfun(a1.x, l1.x), wfun(a1.y, l1.y));
        A.p[3] = __builtin_amdgcn_cvt_pkrtz(wfun(a1.z, l1.z), wfun(a1.w, l1.w));
        A.p[4] = __builtin_amdgcn_cvt_pkrtz(wfun(a2.x, l2.x), wfun(a2.y, l2.y));
        A.p[5] = __builtin_amdgcn_cvt_pkrtz(wfun(a2.z, l2.z), wfun(a2.w, l2.w));
        A.p[6] = __builtin_amdgcn_cvt_pkrtz(wfun(a3.x, l3.x), wfun(a3.y, l3.y));
        A.p[7] = __builtin_amdgcn_cvt_pkrtz(wfun(a3.z, l3.z), wfun(a3.w, l3.w));
        rowsum += rs;

        // ---- B-fragments: two 16B runs per lane (k 8hi.., 16+8hi..) from transposed whT ----
        union { v16h v; float4 q[2]; } B0, B1;
        B0.q[0] = *(const float4*)(pb0 + j0);
        B0.q[1] = *(const float4*)(pb0 + j0 + 16);
        B1.q[0] = *(const float4*)(pb1 + j0);
        B1.q[1] = *(const float4*)(pb1 + j0 + 16);

        c0 = __builtin_amdgcn_wmma_f32_16x16x32_f16(
                 false, A.v, false, B0.v, (short)0, c0, false, false);
        c1 = __builtin_amdgcn_wmma_f32_16x16x32_f16(
                 false, A.v, false, B1.v, (short)0, c1, false, false);

        wait_async_lds();      // our async fills of the next buffer landed
        __syncthreads();       // everyone's fills landed; cur buffer free
    }

    // lanes l and l+16 hold disjoint K subsets of row M: combine, publish
    rowsum += __shfl_xor(rowsum, 16, 32);
    if (lane < 16) s_row[h][M] = rowsum;
    __syncthreads();

    // ---- normalize + store (C layout: VGPR r, lane -> M=r+8*hi, N=col) ----
    #pragma unroll
    for (int r = 0; r < 8; ++r) {
        const int m = r + 8 * hi;
        const float inv = 1.0f / s_row[h][m];
        const size_t o = (size_t)(n0 + m) * OUT_F + h * OPH + col;
        out[o]      = c0[r] * inv;
        out[o + 16] = c1[r] * inv;
    }
}

// ---------------------------------------------------------------------------
extern "C" void kernel_launch(void* const* d_in, const int* in_sizes, int n_in,
                              void* d_out, int out_size, void* d_ws, size_t ws_size,
                              hipStream_t stream)
{
    const float* h_mat  = (const float*)d_in[0];
    const float* adj    = (const float*)d_in[1];
    const float* weight = (const float*)d_in[2];
    const float* att_w  = (const float*)d_in[3];
    float* out = (float*)d_out;

    // workspace layout: whT (128 x N f16) | lp (N*H f32) | lc (N*H f32)
    _Float16* whT = (_Float16*)d_ws;
    float* lp = (float*)((char*)d_ws + (size_t)OUT_F * N_NODES * sizeof(_Float16));
    float* lc = lp + (size_t)N_NODES * H_HEADS;

    gat_prep<<<N_NODES, IN_F, 0, stream>>>(h_mat, weight, att_w, whT, lp, lc);
    gat_main<<<N_NODES / 16, 128, 0, stream>>>(adj, whT, lp, lc, out);
}
